// VanillaRNN_7026566496325
// MI455X (gfx1250) — compile-verified, matrix operand order checked by reading, not run on hardware
//
#include <hip/hip_runtime.h>

#define B_ 512
#define S_ 512
#define H_ 512
#define O_ 10

typedef __attribute__((ext_vector_type(16))) __bf16 v16bf;
typedef __attribute__((ext_vector_type(8)))  __bf16 v8bf;
typedef __attribute__((ext_vector_type(8)))  float  v8f;

static __device__ __forceinline__ __bf16 f32_to_bf16_rne(float f) {
  unsigned u = __float_as_uint(f);
  u += 0x7FFFu + ((u >> 16) & 1u);           // round-to-nearest-even
  unsigned short s = (unsigned short)(u >> 16);
  __bf16 r;
  __builtin_memcpy(&r, &s, sizeof(r));
  return r;
}

static __device__ __forceinline__ float bf16_to_f32(__bf16 h) {
  unsigned short s;
  __builtin_memcpy(&s, &h, sizeof(s));
  unsigned u = ((unsigned)s) << 16;
  return __uint_as_float(u);
}

// Native CDNA5 tanh (V_TANH_F32) instead of the ~20-op OCML expansion.
static __device__ __forceinline__ float fast_tanh(float x) {
#if __has_builtin(__builtin_amdgcn_tanhf)
  return __builtin_amdgcn_tanhf(x);
#elif __has_builtin(__builtin_amdgcn_tanh_f32)
  return __builtin_amdgcn_tanh_f32(x);
#else
  float y;
  asm("v_tanh_f32 %0, %1" : "=v"(y) : "v"(x));
  return y;
#endif
}

// One-time conversion of Whh (f32, row-major [H][H]) to bf16 in workspace.
__global__ void cvt_whh_bf16(const float* __restrict__ whh,
                             __bf16* __restrict__ out, int n) {
  int i = blockIdx.x * blockDim.x + threadIdx.x;
  if (i < n) out[i] = f32_to_bf16_rne(whh[i]);
}

// Persistent per-workgroup RNN: each WG owns 16 batch rows and iterates all
// S steps.  8 waves; wave w computes h_new columns [w*64, w*64+64) via 4
// v_wmma_f32_16x16x32_bf16 tiles per K-block.
__global__ __launch_bounds__(256)
void rnn_persistent(const float* __restrict__ x,
                    const float* __restrict__ whx_w,
                    const float* __restrict__ whx_b,
                    const float* __restrict__ whh_b,
                    const __bf16* __restrict__ whh16,
                    const float* __restrict__ wph_w,
                    const float* __restrict__ wph_b,
                    float* __restrict__ out) {
  __shared__ __bf16 hA[2][16][H_];   // double-buffered h tile, bf16, 32 KB
  __shared__ float  xT[16][S_];      // this WG's x rows, 32 KB

  const int tid  = threadIdx.x;
  const int lane = tid & 31;
  const int wv   = tid >> 5;         // wave id 0..7 (wave32)
  const int ln   = lane & 15;        // N (or M for A-frag) within tile
  const int g    = lane >> 4;        // half-wave selects K group
  const int rowBase = blockIdx.x * 16;

  // Stage x tile: xT[m][t] = x[rowBase+m][t]
  for (int idx = tid; idx < 16 * S_; idx += 256) {
    int m = idx >> 9;
    int t = idx & (S_ - 1);
    xT[m][t] = x[(rowBase + m) * S_ + t];
  }
  // h0 = 0
  {
    int* p = (int*)&hA[0][0][0];
    for (int idx = tid; idx < 16 * H_ / 2; idx += 256) p[idx] = 0;
  }

  // Per-wave column constants: combined bias and Whx column weights
  const int colBase = wv * 64;
  float biasv[4], whxv[4];
#pragma unroll
  for (int tn = 0; tn < 4; ++tn) {
    int n = colBase + tn * 16 + ln;
    biasv[tn] = whx_b[n] + whh_b[n];
    whxv[tn]  = whx_w[n];
  }

  // Warm the caches with this wave's weight slice (re-read every step):
  // 64 rows x 512 cols bf16 -> one prefetch per 64B line segment per lane.
  {
    const char* base = (const char*)(whh16 + (size_t)colBase * H_);
    for (int off = lane * 64; off < 64 * H_ * 2; off += 32 * 64)
      __builtin_prefetch(base + off, 0, 1);
  }

  __syncthreads();

  const v8f vzero = {0.f, 0.f, 0.f, 0.f, 0.f, 0.f, 0.f, 0.f};

#pragma clang loop unroll(disable)
  for (int t = 0; t < S_; ++t) {
    const int cur = t & 1, nxt = cur ^ 1;
    v8f acc[4];
#pragma unroll
    for (int tn = 0; tn < 4; ++tn) acc[tn] = vzero;

#pragma unroll 4
    for (int kb = 0; kb < H_ / 32; ++kb) {
      const int k0 = kb * 32 + g * 8;
      // A fragment (16x32, M=batch rows) from LDS; ISA 16-bit A layout:
      // lane g=0 holds K = k0..k0+7 and k0+16..k0+23; g=1 shifted by 8.
      v8bf alo = *(const v8bf*)&hA[cur][ln][k0];
      v8bf ahi = *(const v8bf*)&hA[cur][ln][k0 + 16];
      v16bf a;
#pragma unroll
      for (int i = 0; i < 8; ++i) { a[i] = alo[i]; a[i + 8] = ahi[i]; }

#pragma unroll
      for (int tn = 0; tn < 4; ++tn) {
        // B fragment: column n of B = row n of Whh (we need h @ Whh^T),
        // same K packing as A (N-major mirror), from L2-resident bf16.
        const __bf16* brow = whh16 + (colBase + tn * 16 + ln) * H_ + k0;
        v8bf blo = *(const v8bf*)brow;
        v8bf bhi = *(const v8bf*)(brow + 16);
        v16bf b;
#pragma unroll
        for (int i = 0; i < 8; ++i) { b[i] = blo[i]; b[i + 8] = bhi[i]; }
        acc[tn] = __builtin_amdgcn_wmma_f32_16x16x32_bf16(
            false, a, false, b, (short)0, acc[tn], false, false);
      }
    }

    // x[b,t] values for the 8 output rows this lane owns (C layout: M=r+8g)
    float xv[8];
#pragma unroll
    for (int r = 0; r < 8; ++r) xv[r] = xT[r + 8 * g][t];

#pragma unroll
    for (int tn = 0; tn < 4; ++tn) {
      const int n = colBase + tn * 16 + ln;
#pragma unroll
      for (int r = 0; r < 8; ++r) {
        const int m = r + 8 * g;
        float hv = fmaf(xv[r], whxv[tn], acc[tn][r] + biasv[tn]);
        hv = fast_tanh(hv);
        hA[nxt][m][n] = f32_to_bf16_rne(hv);
      }
    }
    __syncthreads();   // h_new fully written before next step reads it
  }

  // Final projection: out[16, O] = hT @ Wph^T + Wph_b (tiny, f32 VALU)
  const int fin = S_ & 1;
  if (tid < 16 * O_) {
    const int m = tid / O_;
    const int o = tid - m * O_;
    float s = wph_b[o];
    for (int k = 0; k < H_; ++k)
      s += bf16_to_f32(hA[fin][m][k]) * wph_w[o * H_ + k];
    out[(rowBase + m) * O_ + o] = s;
  }
}

extern "C" void kernel_launch(void* const* d_in, const int* in_sizes, int n_in,
                              void* d_out, int out_size, void* d_ws, size_t ws_size,
                              hipStream_t stream) {
  const float* x     = (const float*)d_in[0];   // [B,S]
  const float* whx_w = (const float*)d_in[1];   // [H,1]
  const float* whx_b = (const float*)d_in[2];   // [H]
  const float* whh_w = (const float*)d_in[3];   // [H,H]
  const float* whh_b = (const float*)d_in[4];   // [H]
  const float* wph_w = (const float*)d_in[5];   // [O,H]
  const float* wph_b = (const float*)d_in[6];   // [O]
  float* out = (float*)d_out;                   // [B,O]

  __bf16* whh16 = (__bf16*)d_ws;                // 512 KB scratch

  cvt_whh_bf16<<<(H_ * H_ + 255) / 256, 256, 0, stream>>>(whh_w, whh16, H_ * H_);
  rnn_persistent<<<B_ / 16, 256, 0, stream>>>(x, whx_w, whx_b, whh_b, whh16,
                                              wph_w, wph_b, out);
}
